// GraphSAGEConv_53618371723538
// MI455X (gfx1250) — matrix operand order repaired and checked by reading, not compiled, not run on hardware
//
#include <hip/hip_runtime.h>

// ---------------------------------------------------------------------------
// Fused batched-GraphConv (3 layers + LN + ELU + linear readout) for gfx1250.
//   B=32768 graphs, 9 nodes/graph, 32 edges/graph, D=128.
//   One workgroup (256 threads, 8 wave32) handles 16 graphs = 144 nodes.
//   - feature tile staged via TDM tensor_load_to_lds (overlapped with degree
//     computation), waited with s_wait_tensorcnt
//   - GEMM h@W with v_wmma_f32_16x16x32_bf16, f32 accumulate
//   - B fragments pre-swizzled in workspace so each is two b128 loads
// ---------------------------------------------------------------------------

#define NBATCH   32768
#define NPER     9
#define EPER     32
#define DIM      128
#define GPB      16                 // graphs per block
#define NODES    (GPB * NPER)       // 144 = 9 m-tiles of 16
#define MTILES   (NODES / 16)       // 9
#define THREADS  256
#define EDGES_WG (GPB * EPER)       // 512

typedef __attribute__((ext_vector_type(16))) __bf16        v16bf;
typedef __attribute__((ext_vector_type(8)))  float         v8f;
typedef __attribute__((ext_vector_type(4)))  unsigned int  v4u;
typedef __attribute__((ext_vector_type(8)))  int           i32x8;
typedef __attribute__((ext_vector_type(4)))  int           i32x4;

union AFrag { v4u q[2]; v16bf v; };                 // 32 B: two 16B LDS loads
union BFrag { v4u q[2]; v16bf v; };                 // 32 B: two 16B global loads
struct U2   { unsigned int x, y; };                 // 4 packed bf16

#if __has_builtin(__builtin_amdgcn_tensor_load_to_lds) && \
    __has_builtin(__builtin_amdgcn_s_wait_tensorcnt)
#define USE_TDM 1
#endif

__device__ __forceinline__ unsigned short f2bf(float f) {
    unsigned int u = __builtin_bit_cast(unsigned int, f);
    unsigned int r = u + 0x7FFFu + ((u >> 16) & 1u);   // round-to-nearest-even
    return (unsigned short)(r >> 16);
}

// --- kernel 0: W0/W1/W2 (128x128 f32, row-major [K][N]) -> bf16, pre-swizzled
// into WMMA B-fragment order:
//   out[ l*16384 + kk*4096 + n*32 + h*16 + e ] = bf16( W_l[(kk*32+h*16+e)*128+n] )
// so lane (n = col, h = lane>>4) reads its 16 elems as two contiguous b128s.
__global__ void wswz_kernel(const float* __restrict__ W0,
                            const float* __restrict__ W1,
                            const float* __restrict__ W2,
                            unsigned short* __restrict__ out) {
    int i  = blockIdx.x * THREADS + threadIdx.x;       // 3*16384 total
    int e  = i & 15;
    int h  = (i >> 4) & 1;
    int n  = (i >> 5) & 127;
    int kk = (i >> 12) & 3;
    int l  = i >> 14;
    const float* W = (l == 0) ? W0 : ((l == 1) ? W1 : W2);
    out[i] = f2bf(W[(kk * 32 + h * 16 + e) * DIM + n]);
}

// --- main fused kernel -----------------------------------------------------
__global__ __launch_bounds__(THREADS)
void gnn_fused_kernel(const float* __restrict__ feat,
                      const int*   __restrict__ src,
                      const int*   __restrict__ dst,
                      const unsigned short* __restrict__ Wbf,   // swizzled [3][16384]
                      const float* __restrict__ b0, const float* __restrict__ g0, const float* __restrict__ bt0,
                      const float* __restrict__ b1, const float* __restrict__ g1, const float* __restrict__ bt1,
                      const float* __restrict__ b2, const float* __restrict__ g2, const float* __restrict__ bt2,
                      const float* __restrict__ Wp, const float* __restrict__ bp,
                      float* __restrict__ out) {
    extern __shared__ char smem[];
    unsigned short* hA  = (unsigned short*)(smem);                    // 144*128 bf16 = 36864 B
    float*          hF  = (float*)(smem + 36864);                     // 144*128 f32  = 73728 B
    unsigned int*   edg = (unsigned int*)(smem + 36864 + 73728);      // 512 u32      =  2048 B
    float*          dOi = (float*)(smem + 112640);                    // 144 f32
    float*          dIi = (float*)(smem + 113216);                    // 144 f32
    int*            cO  = (int*)  (smem + 113792);                    // 144 i32
    int*            cI  = (int*)  (smem + 114368);                    // 144 i32

    const int tid  = threadIdx.x;
    const int lane = tid & 31;
    const int wave = tid >> 5;
    const int G0   = blockIdx.x * GPB;       // first graph of this block
    const int N0   = G0 * NPER;              // first global node

#ifdef USE_TDM
    // ---- kick off TDM: contiguous 144x128 f32 feature tile -> hF ----------
    if (wave == 0) {
        unsigned long long ga = (unsigned long long)(uintptr_t)(feat + (size_t)N0 * DIM);
        unsigned int       la = (unsigned int)(uintptr_t)hF;   // LDS aperture offset
        const unsigned int el = NODES * DIM;                   // 18432 elems of 4 B
        v4u g0v;
        g0v[0] = 1u;                                           // count=1 (valid user D#)
        g0v[1] = la;                                           // lds_addr
        g0v[2] = (unsigned int)ga;                             // global_addr[31:0]
        g0v[3] = (unsigned int)((ga >> 32) & 0x01FFFFFFu)      // global_addr[56:32]
               | (2u << 30);                                   // type=2 (image)
        i32x8 g1v;
        g1v[0] = (int)(2u << 16);                              // data_size=4B, no mask
        g1v[1] = (int)((el & 0xFFFFu) << 16);                  // tensor_dim0[15:0]
        g1v[2] = (int)((el >> 16) | (1u << 16));               // tensor_dim0[31:16] | tensor_dim1=1
        g1v[3] = (int)((el & 0xFFFFu) << 16);                  // tensor_dim1 hi=0 | tile_dim0
        g1v[4] = 1;                                            // tile_dim1=1, tile_dim2=0
        g1v[5] = (int)el;                                      // tensor_dim0_stride[31:0]
        g1v[6] = 0;
        g1v[7] = 0;
        i32x4 z4 = {0, 0, 0, 0};
        i32x8 z8 = {0, 0, 0, 0, 0, 0, 0, 0};
        __builtin_amdgcn_tensor_load_to_lds(g0v, g1v, z4, z4, z8, 0);
    }
#endif

    // ---- phase 0 (overlapped with TDM): edge lists + degrees --------------
    for (int i = tid; i < NODES; i += THREADS) { cO[i] = 0; cI[i] = 0; }
    __syncthreads();
    for (int t = tid; t < EDGES_WG; t += THREADS) {
        int gl   = t >> 5;
        int eidx = (G0 + gl) * EPER + (t & 31);
        int base = (G0 + gl) * NPER;
        int ls   = src[eidx] - base;
        int ld   = dst[eidx] - base;
        edg[t]   = (unsigned int)ls | ((unsigned int)ld << 16);
        atomicAdd(&cO[gl * NPER + ls], 1);
        atomicAdd(&cI[gl * NPER + ld], 1);
    }
    __syncthreads();
    for (int i = tid; i < NODES; i += THREADS) {
        dOi[i] = rsqrtf((float)(cO[i] > 0 ? cO[i] : 1));
        dIi[i] = rsqrtf((float)(cI[i] > 0 ? cI[i] : 1));
    }
#ifdef USE_TDM
    if (wave == 0) __builtin_amdgcn_s_wait_tensorcnt(0);       // hF tile landed
#endif
    __syncthreads();

    // ---- phase 1: features -> bf16 A-tile, pre-scaled by deg_out^-1/2 -----
#ifdef USE_TDM
    for (int t = tid; t < NODES * DIM / 4; t += THREADS) {
        float4 v = *(const float4*)(hF + t * 4);
        float  s = dOi[t >> 5];                      // row = (t*4)/128
        U2 p;
        p.x = (unsigned int)f2bf(v.x * s) | ((unsigned int)f2bf(v.y * s) << 16);
        p.y = (unsigned int)f2bf(v.z * s) | ((unsigned int)f2bf(v.w * s) << 16);
        *(U2*)(hA + t * 4) = p;
    }
#else
    const float4* f4 = (const float4*)(feat + (size_t)N0 * DIM);
    for (int t = tid; t < NODES * DIM / 4; t += THREADS) {
        float4 v = f4[t];
        float  s = dOi[t >> 5];
        U2 p;
        p.x = (unsigned int)f2bf(v.x * s) | ((unsigned int)f2bf(v.y * s) << 16);
        p.y = (unsigned int)f2bf(v.z * s) | ((unsigned int)f2bf(v.w * s) << 16);
        *(U2*)(hA + t * 4) = p;
    }
#endif

    float yp0 = 0.f, yp1 = 0.f;                      // readout partials (2 graphs/wave)

    for (int layer = 0; layer < 3; ++layer) {
        __syncthreads();                             // A-tile ready, hF reads done
        __builtin_prefetch(Wbf + (layer + 1 < 3 ? layer + 1 : layer) * (DIM * DIM), 0, 1);

        // --- B fragments (pre-swizzled): two b128 loads each ---------------
        const int ncol = wave * 16 + (lane & 15);
        const unsigned short* wl =
            Wbf + (size_t)layer * (DIM * DIM) + ncol * 32 + (lane >> 4) * 16;
        BFrag bf[4];
#pragma unroll
        for (int kk = 0; kk < 4; ++kk) {
            bf[kk].q[0] = *(const v4u*)(wl + kk * 4096);
            bf[kk].q[1] = *(const v4u*)(wl + kk * 4096 + 8);
        }

        // --- GEMM: 9 m-tiles, K=128 in 4 steps of 32 -----------------------
        const int mrowA = (lane & 15);
        const int kbA   = (lane >> 4) * 16;          // A: lane<16 -> K {0..15}
        for (int mt = 0; mt < MTILES; ++mt) {
            v8f acc = {0.f, 0.f, 0.f, 0.f, 0.f, 0.f, 0.f, 0.f};
            const unsigned short* arow = hA + (mt * 16 + mrowA) * DIM;
#pragma unroll
            for (int kk = 0; kk < 4; ++kk) {
                AFrag a;
                a.q[0] = *(const v4u*)(arow + kk * 32 + kbA);
                a.q[1] = *(const v4u*)(arow + kk * 32 + kbA + 8);
                acc = __builtin_amdgcn_wmma_f32_16x16x32_bf16(
                        false, a.v, false, bf[kk].v, (short)0, acc, false, false);
            }
            // C/D layout: element e -> row mt*16 + (lane>>4)*8 + e, col = ncol
            const int mbase = mt * 16 + (lane >> 4) * 8;
#pragma unroll
            for (int e = 0; e < 8; ++e)
                hF[(mbase + e) * DIM + ncol] = acc[e];
        }
        __syncthreads();                             // hF complete

        // --- aggregate + deg_in scale + bias + LayerNorm + ELU -------------
        const float* bb = (layer == 0) ? b0  : (layer == 1) ? b1  : b2;
        const float* gg = (layer == 0) ? g0  : (layer == 1) ? g1  : g2;
        const float* bt = (layer == 0) ? bt0 : (layer == 1) ? bt1 : bt2;
        const int d0 = lane * 4;                     // each lane owns 4 dims
        const float4 bia = *(const float4*)(bb + d0);
        const float4 gam = *(const float4*)(gg + d0);
        const float4 bet = *(const float4*)(bt + d0);

        for (int ii = 0; ii < 18; ++ii) {            // 18 nodes per wave = 2 graphs
            const int ln = wave * 18 + ii;
            const int gl = ln / NPER, jn = ln % NPER;
            float ax = 0.f, ay = 0.f, az = 0.f, aw = 0.f;
            const unsigned int* eg = edg + gl * EPER;
            for (int e = 0; e < EPER; ++e) {
                unsigned int pk = eg[e];             // uniform across the wave
                if ((pk >> 16) == (unsigned int)jn) {
                    float4 hv = *(const float4*)(hF + (gl * NPER + (pk & 0xFFFFu)) * DIM + d0);
                    ax += hv.x; ay += hv.y; az += hv.z; aw += hv.w;
                }
            }
            const float di = dIi[ln];
            float vx = ax * di + bia.x, vy = ay * di + bia.y;
            float vz = az * di + bia.z, vw = aw * di + bia.w;
            float s  = vx + vy + vz + vw;
            float sq = vx * vx + vy * vy + vz * vz + vw * vw;
#pragma unroll
            for (int m2 = 16; m2 >= 1; m2 >>= 1) {   // wave32 reduction
                s  += __shfl_xor(s,  m2, 32);
                sq += __shfl_xor(sq, m2, 32);
            }
            const float mu  = s * (1.f / 128.f);
            const float var = sq * (1.f / 128.f) - mu * mu;
            const float rs  = rsqrtf(var + 1e-5f);
            float nx = (vx - mu) * rs * gam.x + bet.x;
            float ny = (vy - mu) * rs * gam.y + bet.y;
            float nz = (vz - mu) * rs * gam.z + bet.z;
            float nw = (vw - mu) * rs * gam.w + bet.w;
            nx = nx > 0.f ? nx : __expf(nx) - 1.f;   // ELU(alpha=1)
            ny = ny > 0.f ? ny : __expf(ny) - 1.f;
            nz = nz > 0.f ? nz : __expf(nz) - 1.f;
            nw = nw > 0.f ? nw : __expf(nw) - 1.f;
            if (layer < 2) {
                // fold next layer's deg_out^-1/2 into the bf16 A-tile store
                const float so = dOi[ln];
                U2 p;
                p.x = (unsigned int)f2bf(nx * so) | ((unsigned int)f2bf(ny * so) << 16);
                p.y = (unsigned int)f2bf(nz * so) | ((unsigned int)f2bf(nw * so) << 16);
                *(U2*)(hA + ln * DIM + d0) = p;
            } else {
                const float4 wp = *(const float4*)(Wp + jn * DIM + d0);
                float p = nx * wp.x + ny * wp.y + nz * wp.z + nw * wp.w;
                if (ii < NPER) yp0 += p; else yp1 += p;
            }
        }
    }

    // ---- readout: reduce per-graph dot products, write y ------------------
#pragma unroll
    for (int m2 = 16; m2 >= 1; m2 >>= 1) {
        yp0 += __shfl_xor(yp0, m2, 32);
        yp1 += __shfl_xor(yp1, m2, 32);
    }
    if (lane == 0) {
        const float bias = bp[0];
        out[G0 + wave * 2 + 0] = yp0 + bias;
        out[G0 + wave * 2 + 1] = yp1 + bias;
    }
}

// ---------------------------------------------------------------------------
extern "C" void kernel_launch(void* const* d_in, const int* in_sizes, int n_in,
                              void* d_out, int out_size, void* d_ws, size_t ws_size,
                              hipStream_t stream) {
    const float* feat = (const float*)d_in[0];
    const int*   src  = (const int*)d_in[1];
    const int*   dst  = (const int*)d_in[2];
    const float* W0   = (const float*)d_in[3];
    const float* b0   = (const float*)d_in[4];
    const float* g0   = (const float*)d_in[5];
    const float* bt0  = (const float*)d_in[6];
    const float* W1   = (const float*)d_in[7];
    const float* b1   = (const float*)d_in[8];
    const float* g1   = (const float*)d_in[9];
    const float* bt1  = (const float*)d_in[10];
    const float* W2   = (const float*)d_in[11];
    const float* b2   = (const float*)d_in[12];
    const float* g2   = (const float*)d_in[13];
    const float* bt2  = (const float*)d_in[14];
    const float* Wp   = (const float*)d_in[15];
    const float* bp   = (const float*)d_in[16];
    float*       y    = (float*)d_out;

    unsigned short* wbf = (unsigned short*)d_ws;     // 3*128*128 bf16 = 96 KB

    static const size_t kSmem = 114944;
    (void)hipFuncSetAttribute((const void*)gnn_fused_kernel,
                              hipFuncAttributeMaxDynamicSharedMemorySize,
                              (int)kSmem);

    wswz_kernel<<<(3 * DIM * DIM) / THREADS, THREADS, 0, stream>>>(W0, W1, W2, wbf);
    gnn_fused_kernel<<<NBATCH / GPB, THREADS, kSmem, stream>>>(
        feat, src, dst, wbf,
        b0, g0, bt0, b1, g1, bt1, b2, g2, bt2,
        Wp, bp, y);
}
